// Seq2SeqNetwork_20263655702852
// MI455X (gfx1250) — matrix operand order repaired and checked by reading, not compiled, not run on hardware
//
#include <hip/hip_runtime.h>
#include <math.h>

// Problem dims (from reference)
#define Bq   256
#define Tq   64
#define Fq   32
#define EUq  256   // encoder units
#define DUq  512   // decoder units = 2*EUq
#define VOCq 2048

static constexpr float GINV = 0.9995003746f; // 1/sqrt(1 + 1e-3) (BN inference scale)

typedef __attribute__((ext_vector_type(16))) __bf16        v16bf;
typedef __attribute__((ext_vector_type(8)))  float         v8f;
typedef __attribute__((ext_vector_type(8)))  unsigned int  v8u;

// ---------------------------------------------------------------------------
// WMMA fragment helpers (layouts per CDNA5 ISA §7.12.2, wave32)
// A 16x32 bf16 row-major: lane L -> row L&15, kb = 8*(L>=16),
//   element e -> k = kb + (e&7) + 16*(e>=8). Adjacent e-pairs are k-adjacent,
//   so load 8 dwords (two contiguous 16B runs) and bit-cast. No half-moves.
// ---------------------------------------------------------------------------
__device__ __forceinline__ v16bf frag_a(const __bf16* __restrict__ A, int ld) {
  int lane = threadIdx.x & 31;
  int m = lane & 15;
  int kb = (lane >> 4) << 3;
  const __bf16* row = A + (size_t)m * ld + kb;
  v8u u;
#pragma unroll
  for (int p = 0; p < 8; ++p) {
    int k = 2 * (p & 3) + ((p & 4) << 2);  // 0,2,4,6, 16,18,20,22
    u[p] = *(const unsigned int*)(row + k);
  }
  return __builtin_bit_cast(v16bf, u);
}

// B operands are weights pre-packed in fragment order (see pack_b_kernel):
// one contiguous, 32B-aligned 16-element run per lane per 32x16 tile.
__device__ __forceinline__ v16bf frag_b_pk(const __bf16* __restrict__ P) {
  return *(const v16bf*)(P + ((threadIdx.x & 31) << 4));
}

__device__ __forceinline__ v8f wmma_bf(v16bf a, v16bf b, v8f c) {
  return __builtin_amdgcn_wmma_f32_16x16x32_bf16(false, a, false, b, (short)0, c,
                                                 false, false);
}

__device__ __forceinline__ float sigmoidf_(float x) { return 1.f / (1.f + __expf(-x)); }

// ---------------------------------------------------------------------------
// Pack f32 weight [K,N] into bf16 WMMA-B fragment order:
//   dst[((kt*Ntiles + nt)*32 + lane)*16 + e] = src[k*N + n]
//   n = nt*16 + (lane&15), k = kt*32 + 8*(lane>=16) + (e&7) + 16*(e>=8)
//   (B layout: lane = col, same k order as frag_a)
// ---------------------------------------------------------------------------
__global__ void __launch_bounds__(256) pack_b_kernel(const float* __restrict__ s,
                                                     __bf16* __restrict__ d,
                                                     int N, int total) {
  int o = blockIdx.x * 256 + threadIdx.x;
  if (o >= total) return;
  int e = o & 15;
  int lane = (o >> 4) & 31;
  int tile = o >> 9;
  int ntiles = N >> 4;
  int kt = tile / ntiles, nt = tile - kt * ntiles;
  int n = (nt << 4) + (lane & 15);
  int k = (kt << 5) + ((lane >> 4) << 3) + (e & 7) + ((e & 8) << 1);
  d[o] = (__bf16)s[(size_t)k * N + n];
}

// ---------------------------------------------------------------------------
// Encoder BatchNorm (inference) -> bf16 activations  xn[B,T,F]
// ---------------------------------------------------------------------------
__global__ void __launch_bounds__(256) bn_enc_kernel(const float* __restrict__ x,
                                                     const float* __restrict__ g,
                                                     const float* __restrict__ be,
                                                     __bf16* __restrict__ xn) {
  int i = blockIdx.x * 256 + threadIdx.x;
  int f = i & (Fq - 1);
  xn[i] = (__bf16)(x[i] * (g[f] * GINV) + be[f]);
}

// ---------------------------------------------------------------------------
// Bi-GRU step, both directions per launch. Fuses x-projection (K-tile 0 over
// Wx, K=F=32) with recurrent projection (K-tiles 1..8 over Wh). z/r gates
// accumulate combined; n-gate keeps x/h parts separate (reset_after).
// grid = 64 blocks x 256 thr = 512 waves = 2 dirs * 16 Mtiles * 16 Utiles.
// Packed-B tile index: kt*Ntiles + nt with Ntiles = 768/16 = 48.
// ---------------------------------------------------------------------------
__global__ void __launch_bounds__(256) gru_step_kernel(
    const __bf16* __restrict__ xn,
    const __bf16* __restrict__ Wxf, const __bf16* __restrict__ Whf,
    const float* __restrict__ bxf, const float* __restrict__ bhf,
    const __bf16* __restrict__ Wxb, const __bf16* __restrict__ Whb,
    const float* __restrict__ bxb, const float* __restrict__ bhb,
    __bf16* __restrict__ outf, __bf16* __restrict__ outb, int t) {
  int wave = blockIdx.x * 8 + (threadIdx.x >> 5);
  int dir = wave >> 8;
  int w = wave & 255;
  int mt = w >> 4, ut = w & 15;
  const __bf16* Wx = dir ? Wxb : Wxf;
  const __bf16* Wh = dir ? Whb : Whf;
  const float*  bx = dir ? bxb : bxf;
  const float*  bh = dir ? bhb : bhf;
  __bf16* out = dir ? outb : outf;
  int tt = dir ? (Tq - 1 - t) : t;  // backward dir consumes reversed input
  int b0 = mt << 4, u0 = ut << 4;
  const int NT = 48;  // 768/16 column tiles

  v8f cz = {}, cr = {}, cnx = {}, cnh = {};
  {  // x-projection K-tile (K = F = 32, single ktile)
    v16bf a = frag_a(xn + ((size_t)b0 * Tq + tt) * Fq, Tq * Fq);
    cz  = wmma_bf(a, frag_b_pk(Wx + ((size_t)(ut) << 9)), cz);
    cr  = wmma_bf(a, frag_b_pk(Wx + ((size_t)(16 + ut) << 9)), cr);
    cnx = wmma_bf(a, frag_b_pk(Wx + ((size_t)(32 + ut) << 9)), cnx);
  }
  if (t > 0) {  // recurrent K-tiles (h0 == 0 -> skip at t==0)
    const __bf16* hp = out + ((size_t)b0 * Tq + (t - 1)) * EUq;
#pragma unroll
    for (int j = 0; j < 8; ++j) {
      v16bf a = frag_a(hp + j * 32, Tq * EUq);
      const __bf16* wb = Wh + ((size_t)(j * NT) << 9);
      cz  = wmma_bf(a, frag_b_pk(wb + ((size_t)(ut) << 9)), cz);
      cr  = wmma_bf(a, frag_b_pk(wb + ((size_t)(16 + ut) << 9)), cr);
      cnh = wmma_bf(a, frag_b_pk(wb + ((size_t)(32 + ut) << 9)), cnh);
    }
  }
  int lane = threadIdx.x & 31;
  int u = u0 + (lane & 15);
  int mb = (lane >> 4) << 3;
  float bxz = bx[u], bxr = bx[EUq + u], bxn = bx[2 * EUq + u];
  float bhz = bh[u], bhr = bh[EUq + u], bhn = bh[2 * EUq + u];
#pragma unroll
  for (int r = 0; r < 8; ++r) {
    int b = b0 + mb + r;
    float hp = (t > 0) ? (float)out[((size_t)b * Tq + t - 1) * EUq + u] : 0.f;
    float zg = sigmoidf_(cz[r] + bxz + bhz);
    float rg = sigmoidf_(cr[r] + bxr + bhr);
    float ng = tanhf(cnx[r] + bxn + rg * (cnh[r] + bhn));
    out[((size_t)b * Tq + t) * EUq + u] = (__bf16)(zg * hp + (1.f - zg) * ng);
  }
}

// ---------------------------------------------------------------------------
// Assemble enc_out[B,T,2U] (backward half time-reversed) and dec_h0.
// ---------------------------------------------------------------------------
__global__ void __launch_bounds__(256) assemble_kernel(const __bf16* __restrict__ outf,
                                                       const __bf16* __restrict__ outb,
                                                       __bf16* __restrict__ enc,
                                                       __bf16* __restrict__ hdec) {
  size_t idx = (size_t)blockIdx.x * 256 + threadIdx.x;
  int u = idx & 511;
  int t = (idx >> 9) & 63;
  int b = (int)(idx >> 15);
  enc[idx] = (u < 256) ? outf[((size_t)b * Tq + t) * EUq + u]
                       : outb[((size_t)b * Tq + (Tq - 1 - t)) * EUq + (u - 256)];
  if (t == Tq - 1) {  // final states = processing step 63 of each direction
    hdec[(size_t)b * DUq + u] =
        (u < 256) ? outf[((size_t)b * Tq + Tq - 1) * EUq + u]
                  : outb[((size_t)b * Tq + Tq - 1) * EUq + (u - 256)];
  }
}

// ---------------------------------------------------------------------------
// Generic bf16 GEMM (packed-B), fp32 out: C[m, coff+n] = A@B + bias.
// Wave computes a 16x64 strip (A fragment reused across 4 WMMAs).
// ---------------------------------------------------------------------------
__global__ void __launch_bounds__(256) gemm_bf16_kernel(
    const __bf16* __restrict__ A, int lda, const __bf16* __restrict__ Bp, int Ntiles,
    const float* __restrict__ bias, float* __restrict__ C, long ldc, long coff,
    int Mtiles, int Nstrips, int Ktiles) {
  int wave = blockIdx.x * (blockDim.x >> 5) + (threadIdx.x >> 5);
  if (wave >= Mtiles * Nstrips) return;
  int mt = wave / Nstrips, ns = wave - mt * Nstrips;
  int m0 = mt << 4, n0 = ns << 6, nt0 = ns << 2;
  v8f c0 = {}, c1 = {}, c2 = {}, c3 = {};
  for (int kt = 0; kt < Ktiles; ++kt) {
    v16bf a = frag_a(A + (size_t)m0 * lda + kt * 32, lda);
    const __bf16* bb = Bp + ((size_t)(kt * Ntiles + nt0) << 9);
    c0 = wmma_bf(a, frag_b_pk(bb), c0);
    c1 = wmma_bf(a, frag_b_pk(bb + 512), c1);
    c2 = wmma_bf(a, frag_b_pk(bb + 1024), c2);
    c3 = wmma_bf(a, frag_b_pk(bb + 1536), c3);
  }
  int lane = threadIdx.x & 31;
  int nl = lane & 15, mb = (lane >> 4) << 3;
#pragma unroll
  for (int r = 0; r < 8; ++r) {
    long row = m0 + mb + r;
    float* crow = C + row * ldc + coff;
    crow[n0 + nl]      = c0[r] + bias[n0 + nl];
    crow[n0 + 16 + nl] = c1[r] + bias[n0 + 16 + nl];
    crow[n0 + 32 + nl] = c2[r] + bias[n0 + 32 + nl];
    crow[n0 + 48 + nl] = c3[r] + bias[n0 + 48 + nl];
  }
}

// ---------------------------------------------------------------------------
// Bahdanau attention + context + decoder-input BN. One workgroup per batch row.
// ---------------------------------------------------------------------------
__global__ void __launch_bounds__(256) attention_kernel(
    const float* __restrict__ q, const float* __restrict__ keys,
    const __bf16* __restrict__ enc, const float* __restrict__ Vv,
    const float* __restrict__ bV, const float* __restrict__ emb,
    const int* __restrict__ targ, const float* __restrict__ gam,
    const float* __restrict__ bet, __bf16* __restrict__ xbn, int t) {
  __shared__ float qs[DUq], vs[DUq], sc[Tq];
  __shared__ float ssum;
  int b = blockIdx.x, tid = threadIdx.x;
  qs[tid]       = q[(size_t)b * DUq + tid];
  qs[256 + tid] = q[(size_t)b * DUq + 256 + tid];
  vs[tid]       = Vv[tid];
  vs[256 + tid] = Vv[256 + tid];
  if (tid < Tq) sc[tid] = 0.f;
  __syncthreads();
  int tp = tid >> 2, part = tid & 3;
  const float* krow = keys + ((size_t)b * Tq + tp) * DUq + part * 128;
  float ps = 0.f;
  for (int j = 0; j < 128; ++j)
    ps += tanhf(krow[j] + qs[part * 128 + j]) * vs[part * 128 + j];
  atomicAdd(&sc[tp], ps);
  __syncthreads();
  if (tid == 0) {  // softmax over 64 (tiny; serial)
    float bv = bV[0], mx = -1e30f;
    for (int i = 0; i < Tq; ++i) { sc[i] += bv; mx = fmaxf(mx, sc[i]); }
    float s = 0.f;
    for (int i = 0; i < Tq; ++i) { sc[i] = __expf(sc[i] - mx); s += sc[i]; }
    ssum = s;
  }
  __syncthreads();
  float inv = 1.f / ssum;
  float c0 = 0.f, c1 = 0.f;
  for (int tt = 0; tt < Tq; ++tt) {
    float w = sc[tt] * inv;
    const __bf16* e = enc + ((size_t)b * Tq + tt) * DUq;
    c0 += w * (float)e[tid];
    c1 += w * (float)e[256 + tid];
  }
  __bf16* xr = xbn + (size_t)b * (2 * DUq);
  xr[tid]       = (__bf16)(c0 * gam[tid] * GINV + bet[tid]);
  xr[256 + tid] = (__bf16)(c1 * gam[256 + tid] * GINV + bet[256 + tid]);
  int tok = targ[b * Tq + t];
  const float* er = emb + (size_t)tok * DUq;
  xr[512 + tid] = (__bf16)(er[tid] * gam[512 + tid] * GINV + bet[512 + tid]);
  xr[768 + tid] = (__bf16)(er[256 + tid] * gam[768 + tid] * GINV + bet[768 + tid]);
}

// ---------------------------------------------------------------------------
// Decoder gate GEMM + fused gates: xp = xbn@Wx_d + bx_d; h_prev == 0 in the
// reference, so hp == bh_d and h_new = (1-zg)*ng. Writes hdec (bf16).
// Packed Wxd: Ntiles = 1536/16 = 96; gate chunk g at nt = g*32 + ut.
// grid = 64 blocks x 256 thr = 512 waves = 16 Mtiles * 32 Utiles.
// ---------------------------------------------------------------------------
__global__ void __launch_bounds__(256) dec_gates_kernel(
    const __bf16* __restrict__ xbn, const __bf16* __restrict__ Wxd,
    const float* __restrict__ bxd, const float* __restrict__ bhd,
    __bf16* __restrict__ hdec) {
  int wave = blockIdx.x * 8 + (threadIdx.x >> 5);
  int mt = wave >> 5, ut = wave & 31;
  int b0 = mt << 4, u0 = ut << 4;
  const int NT = 96;
  v8f cz = {}, cr = {}, cn = {};
  for (int kt = 0; kt < 32; ++kt) {  // K = 1024
    v16bf a = frag_a(xbn + (size_t)b0 * 1024 + kt * 32, 1024);
    const __bf16* wb = Wxd + ((size_t)(kt * NT) << 9);
    cz = wmma_bf(a, frag_b_pk(wb + ((size_t)(ut) << 9)), cz);
    cr = wmma_bf(a, frag_b_pk(wb + ((size_t)(32 + ut) << 9)), cr);
    cn = wmma_bf(a, frag_b_pk(wb + ((size_t)(64 + ut) << 9)), cn);
  }
  int lane = threadIdx.x & 31;
  int u = u0 + (lane & 15);
  int mb = (lane >> 4) << 3;
  float bz  = bxd[u] + bhd[u];
  float br  = bxd[DUq + u] + bhd[DUq + u];
  float bn_ = bxd[2 * DUq + u];
  float bhn = bhd[2 * DUq + u];
#pragma unroll
  for (int r = 0; r < 8; ++r) {
    int b = b0 + mb + r;
    float zg = sigmoidf_(cz[r] + bz);
    float rg = sigmoidf_(cr[r] + br);
    float ng = tanhf(cn[r] + bn_ + rg * bhn);
    hdec[(size_t)b * DUq + u] = (__bf16)((1.f - zg) * ng);
  }
}

// ---------------------------------------------------------------------------
extern "C" void kernel_launch(void* const* d_in, const int* in_sizes, int n_in,
                              void* d_out, int out_size, void* d_ws, size_t ws_size,
                              hipStream_t stream) {
  (void)in_sizes; (void)n_in; (void)out_size; (void)ws_size;
  const float* enc_input  = (const float*)d_in[0];
  const float* bn_e_gamma = (const float*)d_in[1];
  const float* bn_e_beta  = (const float*)d_in[2];
  const float* Wx_f = (const float*)d_in[3];
  const float* Wh_f = (const float*)d_in[4];
  const float* bx_f = (const float*)d_in[5];
  const float* bh_f = (const float*)d_in[6];
  const float* Wx_b = (const float*)d_in[7];
  const float* Wh_b = (const float*)d_in[8];
  const float* bx_b = (const float*)d_in[9];
  const float* bh_b = (const float*)d_in[10];
  const float* W1   = (const float*)d_in[11];
  const float* b1   = (const float*)d_in[12];
  const float* W2   = (const float*)d_in[13];
  const float* b2   = (const float*)d_in[14];
  const float* Vv   = (const float*)d_in[15];
  const float* bV   = (const float*)d_in[16];
  const float* emb  = (const float*)d_in[17];
  const float* bn_d_gamma = (const float*)d_in[18];
  const float* bn_d_beta  = (const float*)d_in[19];
  const float* Wx_d = (const float*)d_in[20];
  // d_in[21] = Wh_d: dead in the reference (decoder GRU h_prev == 0)
  const float* bx_d = (const float*)d_in[22];
  const float* bh_d = (const float*)d_in[23];
  const float* Wfc  = (const float*)d_in[24];
  const float* bfc  = (const float*)d_in[25];
  const int*   targ = (const int*)d_in[26];
  float* out = (float*)d_out;

  // Workspace carve-out (~77 MB total, everything L2-resident on 192MB L2)
  char* p = (char*)d_ws;
  auto abf = [&](size_t n) { __bf16* r = (__bf16*)p; p += ((n * 2 + 255) & ~(size_t)255); return r; };
  auto af32 = [&](size_t n) { float* r = (float*)p; p += ((n * 4 + 255) & ~(size_t)255); return r; };
  __bf16* Wxf_h = abf((size_t)Fq * 3 * EUq);
  __bf16* Whf_h = abf((size_t)EUq * 3 * EUq);
  __bf16* Wxb_h = abf((size_t)Fq * 3 * EUq);
  __bf16* Whb_h = abf((size_t)EUq * 3 * EUq);
  __bf16* W1_h  = abf((size_t)DUq * DUq);
  __bf16* W2_h  = abf((size_t)DUq * DUq);
  __bf16* Wxd_h = abf((size_t)(2 * DUq) * 3 * DUq);
  __bf16* Wfc_h = abf((size_t)DUq * VOCq);
  __bf16* xn_h  = abf((size_t)Bq * Tq * Fq);
  __bf16* outf  = abf((size_t)Bq * Tq * EUq);
  __bf16* outb  = abf((size_t)Bq * Tq * EUq);
  __bf16* encO  = abf((size_t)Bq * Tq * DUq);
  __bf16* hdec  = abf((size_t)Bq * DUq);
  __bf16* xbn   = abf((size_t)Bq * 2 * DUq);
  float*  keys  = af32((size_t)Bq * Tq * DUq);
  float*  qbuf  = af32((size_t)Bq * DUq);

  // Pack all GEMM B-operands into WMMA fragment order (bf16)
  auto pack = [&](const float* s, __bf16* d, int K, int N) {
    int total = K * N;
    pack_b_kernel<<<(total + 255) / 256, 256, 0, stream>>>(s, d, N, total);
  };
  pack(Wx_f, Wxf_h, Fq, 3 * EUq);
  pack(Wh_f, Whf_h, EUq, 3 * EUq);
  pack(Wx_b, Wxb_h, Fq, 3 * EUq);
  pack(Wh_b, Whb_h, EUq, 3 * EUq);
  pack(W1, W1_h, DUq, DUq);
  pack(W2, W2_h, DUq, DUq);
  pack(Wx_d, Wxd_h, 2 * DUq, 3 * DUq);
  pack(Wfc, Wfc_h, DUq, VOCq);

  // Encoder: BN -> bf16 activations
  bn_enc_kernel<<<(Bq * Tq * Fq) / 256, 256, 0, stream>>>(enc_input, bn_e_gamma,
                                                          bn_e_beta, xn_h);
  // Bi-GRU over T steps (both directions per launch)
  for (int t = 0; t < Tq; ++t)
    gru_step_kernel<<<64, 256, 0, stream>>>(xn_h, Wxf_h, Whf_h, bx_f, bh_f,
                                            Wxb_h, Whb_h, bx_b, bh_b, outf, outb, t);
  // enc_out + dec_h0
  assemble_kernel<<<(Bq * Tq * DUq) / 256, 256, 0, stream>>>(outf, outb, encO, hdec);
  // keys = enc_out @ W1 + b1   ([16384,512]x[512,512])
  gemm_bf16_kernel<<<1024, 256, 0, stream>>>(encO, DUq, W1_h, DUq / 16, b1, keys,
                                             DUq, 0, (Bq * Tq) / 16, DUq / 64, DUq / 32);
  // Decoder: 64 sequential steps (only dependency: q <- previous h_new)
  for (int t = 0; t < Tq; ++t) {
    gemm_bf16_kernel<<<16, 256, 0, stream>>>(hdec, DUq, W2_h, DUq / 16, b2, qbuf,
                                             DUq, 0, Bq / 16, DUq / 64, DUq / 32);
    attention_kernel<<<Bq, 256, 0, stream>>>(qbuf, keys, encO, Vv, bV, emb, targ,
                                             bn_d_gamma, bn_d_beta, xbn, t);
    dec_gates_kernel<<<64, 256, 0, stream>>>(xbn, Wxd_h, bx_d, bh_d, hdec);
    gemm_bf16_kernel<<<64, 256, 0, stream>>>(hdec, DUq, Wfc_h, VOCq / 16, bfc, out,
                                             (long)Tq * VOCq, (long)t * VOCq,
                                             Bq / 16, VOCq / 64, DUq / 32);
  }
}
